// MultiHeadSelfAttention_67379446940074
// MI455X (gfx1250) — compile-verified, matrix operand order checked by reading, not run on hardware
//
#include <hip/hip_runtime.h>

// ---------------------------------------------------------------------------
// MI455X (gfx1250) multi-head self-attention.
// Compute-bound (~77 GFLOP vs ~38 MB) -> all matmuls on v_wmma_f32_16x16x32_f16
// (f16 in, f32 accumulate). Flash-style online softmax (no [B,H,N,N] tensor).
// Attention K/V tiles are staged into LDS once per block (4 waves share them)
// with double-buffered GLOBAL_LOAD_ASYNC_TO_LDS_B128 (ASYNCcnt) so the next
// tile streams in while the current chunk's WMMAs run.
// ---------------------------------------------------------------------------

typedef __attribute__((ext_vector_type(16))) _Float16 v16h;
typedef __attribute__((ext_vector_type(8)))  _Float16 v8h;
typedef __attribute__((ext_vector_type(8)))  float    v8f;

constexpr int BB  = 2;
constexpr int NN  = 4096;
constexpr int DD  = 512;
constexpr int HH  = 8;
constexpr int HD  = 64;
constexpr int MR  = BB * NN;      // 8192 token rows

static __device__ __forceinline__
v8f wmma16(v16h a, v16h b, v8f c) {
  // D = A(16x32) x B(32x16) + C, f32 accumulate
  return __builtin_amdgcn_wmma_f32_16x16x32_f16(
      /*neg_a=*/false, a, /*neg_b=*/false, b,
      /*c_mod=*/(short)0, c, /*reuse_a=*/false, /*reuse_b=*/false);
}

// Load one 16x32 fragment (works for global or LDS pointers; addrspace is
// inferred after inlining). A operand: base = matrix rows, row0 = M origin.
// B operand (symmetric striping: lane n holds column n of B == row n of B^T):
// base = B^T rows, row0 = N origin. Per lane: row row0+(lane&15), K elements
// {k0..k0+7} and {k0+16..k0+23}, shifted +8 for lanes 16..31.
static __device__ __forceinline__
v16h load_frag(const _Float16* __restrict__ base, int ld, int row0, int k0) {
  const int lane = threadIdx.x & 31;
  const int r  = lane & 15;
  const int hi = lane >> 4;
  const _Float16* p = base + (size_t)(row0 + r) * ld + k0 + hi * 8;
  v8h lo = *(const v8h*)(p);
  v8h hh = *(const v8h*)(p + 16);
  v16h out;
#pragma unroll
  for (int i = 0; i < 8; ++i) { out[i] = lo[i]; out[i + 8] = hh[i]; }
  return out;
}

// CDNA5 async copy: one 16B segment per lane, global -> LDS, ASYNCcnt-tracked.
// vdst = per-lane LDS byte offset (low 32 bits of generic pointer), vaddr =
// 64-bit global address, GV mode.
static __device__ __forceinline__
void async_cp16(const void* lds_dst, const void* gsrc) {
  uint32_t lds_off = (uint32_t)(uintptr_t)lds_dst;
  uint64_t gaddr   = (uint64_t)(uintptr_t)gsrc;
  asm volatile("global_load_async_to_lds_b128 %0, %1, off"
               :: "v"(lds_off), "v"(gaddr) : "memory");
}

// ---------------------------------------------------------------------------
// f32 -> f16 conversion
// ---------------------------------------------------------------------------
__global__ void cvt_f32_to_f16(const float* __restrict__ src,
                               _Float16* __restrict__ dst, int n) {
  int i = blockIdx.x * blockDim.x + threadIdx.x;
  if (i < n) dst[i] = (_Float16)src[i];
}

// ---------------------------------------------------------------------------
// QKV projection: Y = x @ W^T + b.  One wave computes a 16(M) x 64(N) tile.
// grid = (MR/64, DD/64, 3); block = 128 (4 waves).
// z=0 -> Q (natural [MR,DD] f16), z=1 -> K (natural), z=2 -> V transposed
// into vt[B][H][HD][NN] so attention's P.V B-fragments are row reads.
// ---------------------------------------------------------------------------
__global__ void qkv_proj_kernel(const _Float16* __restrict__ xh,
                                const _Float16* __restrict__ wq,
                                const _Float16* __restrict__ wk,
                                const _Float16* __restrict__ wv,
                                const float* __restrict__ bq,
                                const float* __restrict__ bk,
                                const float* __restrict__ bv,
                                _Float16* __restrict__ qh,
                                _Float16* __restrict__ kh,
                                _Float16* __restrict__ vt) {
  const int lane = threadIdx.x & 31;
  const int wave = threadIdx.x >> 5;
  const int mt = blockIdx.x * 4 + wave;   // 0..511
  const int nt = blockIdx.y;              // 0..7
  const int z  = blockIdx.z;              // 0=Q 1=K 2=V
  const _Float16* W    = (z == 0) ? wq : (z == 1) ? wk : wv;
  const float*    bias = (z == 0) ? bq : (z == 1) ? bk : bv;
  const int m0 = mt * 16, n0 = nt * 64;

  v8f acc[4] = {};
  for (int kc = 0; kc < DD; kc += 32) {
    v16h a = load_frag(xh, DD, m0, kc);
#pragma unroll
    for (int j = 0; j < 4; ++j) {
      v16h bf = load_frag(W, DD, n0 + j * 16, kc);   // W rows = (W^T) columns
      acc[j] = wmma16(a, bf, acc[j]);
    }
  }

  const int r  = lane & 15;
  const int hi = lane >> 4;
  if (z < 2) {
    _Float16* out = (z == 0) ? qh : kh;
#pragma unroll
    for (int j = 0; j < 4; ++j) {
      const float bcol = bias[n0 + j * 16 + r];
#pragma unroll
      for (int i = 0; i < 8; ++i) {
        const int row = m0 + i + 8 * hi;
        out[(size_t)row * DD + n0 + j * 16 + r] = (_Float16)(acc[j][i] + bcol);
      }
    }
  } else {
#pragma unroll
    for (int j = 0; j < 4; ++j) {
      const int d = n0 + j * 16 + r;
      const float bcol = bias[d];
      const int h  = d >> 6;
      const int hd = d & 63;
#pragma unroll
      for (int i = 0; i < 8; ++i) {
        const int row = m0 + i + 8 * hi;
        const int b = row >> 12;          // /4096
        const int n = row & 4095;
        vt[(((size_t)b * HH + h) * HD + hd) * NN + n] = (_Float16)(acc[j][i] + bcol);
      }
    }
  }
}

// ---------------------------------------------------------------------------
// Flash attention. Block = 4 waves sharing one (b,h); each wave owns 16 query
// rows. Per 32-key chunk: K tile (32x64, 4KB) and V^T tile (64x32, 4KB) are
// streamed into LDS with double-buffered async copies; each wave then runs
// 4 score WMMAs + 1 row-sum WMMA (all-ones B) + 4 P.V WMMAs from LDS.
// grid = (NN/64, BB*HH); block = 128.
// ---------------------------------------------------------------------------
__global__ void attn_kernel(const _Float16* __restrict__ qh,
                            const _Float16* __restrict__ kh,
                            const _Float16* __restrict__ vt,
                            _Float16* __restrict__ ctx) {
  const int lane = threadIdx.x & 31;
  const int wave = threadIdx.x >> 5;
  const int tid  = threadIdx.x;           // 0..127
  const int qt = blockIdx.x * 4 + wave;   // 0..255
  const int bh = blockIdx.y;              // 0..15
  const int b = bh >> 3, h = bh & 7;

  const _Float16* Qb = qh + (size_t)b * NN * DD;
  const _Float16* Kb = kh + (size_t)b * NN * DD;
  const _Float16* Vb = vt + (size_t)bh * HD * NN;   // rows = hd, ld = NN
  const int q0 = qt * 16;

  __shared__ _Float16 Kt[2][32][64];   // [buf][key][hd]   8 KB
  __shared__ _Float16 Vt[2][64][32];   // [buf][hd][key]   8 KB
  __shared__ _Float16 P[4][16][40];    // per-wave prob tile, 16B-aligned rows

  // Q fragments, prescaled by 1/sqrt(HD) = 0.125 (exact in f16)
  v16h qa0 = load_frag(Qb, DD, q0, h * HD + 0);
  v16h qa1 = load_frag(Qb, DD, q0, h * HD + 32);
#pragma unroll
  for (int i = 0; i < 16; ++i) {
    qa0[i] = qa0[i] * (_Float16)0.125f;
    qa1[i] = qa1[i] * (_Float16)0.125f;
  }
  v16h ones;
#pragma unroll
  for (int i = 0; i < 16; ++i) ones[i] = (_Float16)1.0f;

  float rowM[8];
#pragma unroll
  for (int i = 0; i < 8; ++i) rowM[i] = -1e30f;
  v8f lacc = {};
  v8f o[4] = {};

  _Float16 (*Pw)[40] = P[wave];
  const int r  = lane & 15;
  const int hi = lane >> 4;

  // stage key-chunk kt into buffer `buf`: 4 async b128 instrs per wave
  auto stage = [&](int kt, int buf) {
#pragma unroll
    for (int s = 0; s < 2; ++s) {
      // K tile: 256 x 16B segments, [key][hd] rows of 128B
      int seg = tid + s * 128;
      int row = seg >> 3, c = seg & 7;
      async_cp16(&Kt[buf][row][c * 8],
                 Kb + (size_t)(kt + row) * DD + h * HD + c * 8);
      // V^T tile: 256 x 16B segments, [hd][key] rows of 64B
      int hd = seg >> 2, cv = seg & 3;
      async_cp16(&Vt[buf][hd][cv * 8],
                 Vb + (size_t)hd * NN + kt + cv * 8);
    }
  };

  constexpr int NCH = NN / 32;   // 128 chunks
  stage(0, 0);

  for (int ic = 0; ic < NCH; ++ic) {
    const int buf = ic & 1;
    asm volatile("s_wait_asynccnt 0x0" ::: "memory");  // my chunk-ic copies done
    __syncthreads();                                   // whole tile visible; prev
                                                       // buffer free to overwrite
    if (ic + 1 < NCH) stage((ic + 1) * 32, buf ^ 1);

    const _Float16* Kl = &Kt[buf][0][0];
    const _Float16* Vl = &Vt[buf][0][0];

    // scores: S(16q x 32k) via two 16x16 key sub-tiles, K-dim = HD in 2 chunks
    v16h kb0a = load_frag(Kl, HD, 0,  0);
    v16h kb0b = load_frag(Kl, HD, 0,  32);
    v16h kb1a = load_frag(Kl, HD, 16, 0);
    v16h kb1b = load_frag(Kl, HD, 16, 32);
    v8f s0 = {}, s1 = {};
    s0 = wmma16(qa0, kb0a, s0);  s0 = wmma16(qa1, kb0b, s0);
    s1 = wmma16(qa0, kb1a, s1);  s1 = wmma16(qa1, kb1b, s1);

    // online softmax: row max over 16 lanes of each half-wave (row = vgpr idx)
    float t[8];
#pragma unroll
    for (int i = 0; i < 8; ++i) t[i] = fmaxf(s0[i], s1[i]);
#pragma unroll
    for (int m = 1; m <= 8; m <<= 1) {
#pragma unroll
      for (int i = 0; i < 8; ++i) t[i] = fmaxf(t[i], __shfl_xor(t[i], m, 32));
    }
    float alpha[8];
#pragma unroll
    for (int i = 0; i < 8; ++i) {
      const float mn = fmaxf(rowM[i], t[i]);
      alpha[i] = __expf(rowM[i] - mn);
      rowM[i] = mn;
    }

    // P = exp(S - rowmax) -> LDS (accumulator layout: row = i+8*hi, col = r)
#pragma unroll
    for (int i = 0; i < 8; ++i) {
      const int row = i + 8 * hi;
      Pw[row][r]      = (_Float16)__expf(s0[i] - rowM[i]);
      Pw[row][r + 16] = (_Float16)__expf(s1[i] - rowM[i]);
    }
    asm volatile("s_wait_dscnt 0x0" ::: "memory");  // per-wave LDS RAW (in-order)

    // reload P as an A fragment (16q x 32k)
    v16h pf;
    {
      const _Float16* pp = &Pw[r][hi * 8];
      v8h lo = *(const v8h*)(pp);
      v8h hh = *(const v8h*)(pp + 16);
#pragma unroll
      for (int i = 0; i < 8; ++i) { pf[i] = lo[i]; pf[i + 8] = hh[i]; }
    }

    // rescale running sum + context by alpha, then accumulate
#pragma unroll
    for (int i = 0; i < 8; ++i) lacc[i] *= alpha[i];
#pragma unroll
    for (int j = 0; j < 4; ++j)
#pragma unroll
      for (int i = 0; i < 8; ++i) o[j][i] *= alpha[i];

    lacc = wmma16(pf, ones, lacc);                  // row sums via WMMA
#pragma unroll
    for (int j = 0; j < 4; ++j) {
      v16h vb = load_frag(Vl, 32, j * 16, 0);       // V^T rows -> B fragment
      o[j] = wmma16(pf, vb, o[j]);
    }
  }

  // normalize and store context (natural [B,N,D] f16, heads merged)
  float inv[8];
#pragma unroll
  for (int i = 0; i < 8; ++i) inv[i] = 1.0f / lacc[i];
  _Float16* Cb = ctx + ((size_t)b * NN + q0) * DD + h * HD;
#pragma unroll
  for (int j = 0; j < 4; ++j)
#pragma unroll
    for (int i = 0; i < 8; ++i) {
      const int row = i + 8 * hi;
      Cb[(size_t)row * DD + j * 16 + r] = (_Float16)(o[j][i] * inv[i]);
    }
}

// ---------------------------------------------------------------------------
// Output projection: out = ctx @ Wo^T + bo, f32 result.
// grid = (MR/64, DD/64); block = 128.
// ---------------------------------------------------------------------------
__global__ void out_proj_kernel(const _Float16* __restrict__ ctx,
                                const _Float16* __restrict__ woh,
                                const float* __restrict__ bo,
                                float* __restrict__ out) {
  const int lane = threadIdx.x & 31;
  const int wave = threadIdx.x >> 5;
  const int mt = blockIdx.x * 4 + wave;
  const int nt = blockIdx.y;
  const int m0 = mt * 16, n0 = nt * 64;

  v8f acc[4] = {};
  for (int kc = 0; kc < DD; kc += 32) {
    v16h a = load_frag(ctx, DD, m0, kc);
#pragma unroll
    for (int j = 0; j < 4; ++j) {
      v16h bf = load_frag(woh, DD, n0 + j * 16, kc);
      acc[j] = wmma16(a, bf, acc[j]);
    }
  }
  const int r  = lane & 15;
  const int hi = lane >> 4;
#pragma unroll
  for (int j = 0; j < 4; ++j) {
    const float bcol = bo[n0 + j * 16 + r];
#pragma unroll
    for (int i = 0; i < 8; ++i) {
      const int row = m0 + i + 8 * hi;
      out[(size_t)row * DD + n0 + j * 16 + r] = acc[j][i] + bcol;
    }
  }
}

// ---------------------------------------------------------------------------
extern "C" void kernel_launch(void* const* d_in, const int* in_sizes, int n_in,
                              void* d_out, int out_size, void* d_ws, size_t ws_size,
                              hipStream_t stream) {
  const float* x  = (const float*)d_in[0];
  const float* Wq = (const float*)d_in[1];
  const float* bq = (const float*)d_in[2];
  const float* Wk = (const float*)d_in[3];
  const float* bk = (const float*)d_in[4];
  const float* Wv = (const float*)d_in[5];
  const float* bv = (const float*)d_in[6];
  const float* Wo = (const float*)d_in[7];
  const float* bo = (const float*)d_in[8];
  float* out = (float*)d_out;

  // workspace carve-up (f16 buffers), total ~42 MB
  const size_t NXH = (size_t)MR * DD;          // 4,194,304
  const size_t NW  = (size_t)DD * DD;          //   262,144
  char* ws = (char*)d_ws;
  _Float16* xh  = (_Float16*)ws;                    ws += NXH * 2;
  _Float16* wqh = (_Float16*)ws;                    ws += NW * 2;
  _Float16* wkh = (_Float16*)ws;                    ws += NW * 2;
  _Float16* wvh = (_Float16*)ws;                    ws += NW * 2;
  _Float16* woh = (_Float16*)ws;                    ws += NW * 2;
  _Float16* qh  = (_Float16*)ws;                    ws += NXH * 2;
  _Float16* kh  = (_Float16*)ws;                    ws += NXH * 2;
  _Float16* vt  = (_Float16*)ws;                    ws += NXH * 2;
  _Float16* ctx = (_Float16*)ws;                    ws += NXH * 2;

  // 1) convert inputs to f16
  cvt_f32_to_f16<<<(int)(NXH / 256), 256, 0, stream>>>(x,  xh,  (int)NXH);
  cvt_f32_to_f16<<<(int)(NW  / 256), 256, 0, stream>>>(Wq, wqh, (int)NW);
  cvt_f32_to_f16<<<(int)(NW  / 256), 256, 0, stream>>>(Wk, wkh, (int)NW);
  cvt_f32_to_f16<<<(int)(NW  / 256), 256, 0, stream>>>(Wv, wvh, (int)NW);
  cvt_f32_to_f16<<<(int)(NW  / 256), 256, 0, stream>>>(Wo, woh, (int)NW);

  // 2) QKV projections (V transposed)
  qkv_proj_kernel<<<dim3(MR / 64, DD / 64, 3), 128, 0, stream>>>(
      xh, wqh, wkh, wvh, bq, bk, bv, qh, kh, vt);

  // 3) flash attention (async double-buffered K/V staging in LDS)
  attn_kernel<<<dim3(NN / 64, BB * HH), 128, 0, stream>>>(qh, kh, vt, ctx);

  // 4) output projection
  out_proj_kernel<<<dim3(MR / 64, DD / 64), 128, 0, stream>>>(ctx, woh, bo, out);
}
